// MultiheadAttentionBlock_9749575762159
// MI455X (gfx1250) — compile-verified, matrix operand order checked by reading, not run on hardware
//
#include <hip/hip_runtime.h>

// MI455X / gfx1250, wave32. Full MHA block on the f16 WMMA path
// (v_wmma_f32_16x16x32_f16), f32 accumulation, flash-attention softmax.
// Round 3: add __launch_bounds__ + shorter live ranges to eliminate the
// scratch spills seen in round 2 while keeping the big register tiles:
//   - GEMM wave tile 32(M) x 64(N): 8 WMMAs per 12 b128 loads (~43 F/B)
//   - Attention: 32 queries/wave: 16 WMMAs per 16 b128 loads (~64 F/B)

typedef __attribute__((ext_vector_type(16))) _Float16 v16h;
typedef __attribute__((ext_vector_type(8)))  _Float16 v8h;
typedef __attribute__((ext_vector_type(4)))  _Float16 v4h;
typedef __attribute__((ext_vector_type(8)))  float    v8f;
typedef __attribute__((ext_vector_type(4)))  float    v4f;

#define D_MODEL 1024
#define SEQ     2048
#define BATCH   2
#define NHEAD   16
#define DK      64

// ---------------------------------------------------------------------------
// A/B operand loader for V_WMMA_F32_16X16X32_F16, NT-form.
// Source matrix is row-major [rows, ld] (rows = M for A, rows = N for B,
// columns = K). Per the CDNA5 ISA 16-bit A layout (CDNA5 ISA §7.12.2):
//   lane L holds row (L&15); VGPRs 0..3 hold K = koff..koff+7 and
//   VGPRs 4..7 hold K = 16+koff..16+koff+7, where koff = 8*(L>>4).
// Both 8-half chunks are contiguous -> two 16B vector loads per lane.
// ---------------------------------------------------------------------------
__device__ __forceinline__ v16h load_tile_nt(const _Float16* __restrict__ base,
                                             int row, int ld, int kbase, int lane) {
  const int r    = row + (lane & 15);
  const int koff = (lane >> 4) * 8;
  const _Float16* p = base + (size_t)r * ld + (kbase + koff);
  v8h lo = *(const v8h*)(p);
  v8h hi = *(const v8h*)(p + 16);
  v16h out;
#pragma unroll
  for (int i = 0; i < 8; ++i) { out[i] = lo[i]; out[i + 8] = hi[i]; }
  return out;
}

__device__ __forceinline__ v8f wmma_f16(v16h a, v16h b, v8f c) {
  // D = A(16x32 f16) * B(32x16 f16) + C(16x16 f32)
  return __builtin_amdgcn_wmma_f32_16x16x32_f16(
      /*neg_a=*/false, a, /*neg_b=*/false, b,
      /*c_mod=*/(short)0, c, /*reuse_a=*/false, /*reuse_b=*/false);
}

// ---------------------------------------------------------------------------
// f32 -> f16 conversion, 4 elements / thread.
// ---------------------------------------------------------------------------
__global__ void __launch_bounds__(256)
cvt_f32_to_f16_kernel(const float* __restrict__ in,
                      _Float16* __restrict__ out, int n4) {
  int i = blockIdx.x * blockDim.x + threadIdx.x;
  if (i < n4) {
    v4f v = ((const v4f*)in)[i];
    v4h h;
#pragma unroll
    for (int j = 0; j < 4; ++j) h[j] = (_Float16)v[j];
    ((v4h*)out)[i] = h;
  }
}

// ---------------------------------------------------------------------------
// NT GEMM: Y[M,N] = A[M,K] @ W[N,K]^T + bias, M=4096, N=K=1024.
// Block = 256 threads = 8 waves; each wave owns a 32(M) x 64(N) tile
// (2 A-tiles x 4 B-tiles -> 8 WMMAs per k-step, B reused across M).
// MODE 0: f16 row-major out    (Q, K projections)
// MODE 1: f16 transposed-per-head out Vt[b,h,d,s]  (V projection)
// MODE 2: f32 row-major out + bias (final output projection)
// ---------------------------------------------------------------------------
template <int MODE>
__global__ void __launch_bounds__(256)
gemm_nt_kernel(const _Float16* __restrict__ A,
               const _Float16* __restrict__ W,
               const float* __restrict__ bias,
               void* __restrict__ Yv) {
  const int lane  = threadIdx.x & 31;
  const int wave  = threadIdx.x >> 5;
  const int mbase = blockIdx.x * 32;
  const int nbase = blockIdx.y * 512 + wave * 64;

  v8f acc[2][4] = {};
  for (int k = 0; k < D_MODEL; k += 32) {
    v16h a0 = load_tile_nt(A, mbase,      D_MODEL, k, lane);
    v16h a1 = load_tile_nt(A, mbase + 16, D_MODEL, k, lane);
#pragma unroll
    for (int t = 0; t < 4; ++t) {
      v16h b = load_tile_nt(W, nbase + 16 * t, D_MODEL, k, lane);
      acc[0][t] = wmma_f16(a0, b, acc[0][t]);
      acc[1][t] = wmma_f16(a1, b, acc[1][t]);
    }
  }

  // C/D layout: lane -> N = nbase+16t+(lane&15); VGPR v -> M = +v+8*(lane>>4)
  const int hi = lane >> 4;
  const int ln = lane & 15;
#pragma unroll
  for (int mt = 0; mt < 2; ++mt) {
#pragma unroll
    for (int t = 0; t < 4; ++t) {
      const int   n  = nbase + 16 * t + ln;
      const float bv = bias[n];
#pragma unroll
      for (int v = 0; v < 8; ++v) {
        const int   m = mbase + 16 * mt + v + 8 * hi;
        const float y = acc[mt][t][v] + bv;
        if (MODE == 0) {
          ((_Float16*)Yv)[(size_t)m * D_MODEL + n] = (_Float16)y;
        } else if (MODE == 1) {
          const int b = m >> 11, s = m & (SEQ - 1);
          const int h = n >> 6,  d = n & (DK - 1);
          ((_Float16*)Yv)[((((size_t)b * NHEAD + h) * DK + d) << 11) + s] = (_Float16)y;
        } else {
          ((float*)Yv)[(size_t)m * D_MODEL + n] = y;
        }
      }
    }
  }
}

// ---------------------------------------------------------------------------
// Online-softmax update for one 16-query group.
// s0/s1: S^T score tiles (keys kb..+15 / kb+16..+31; lane = query, VGPR = key).
// Rescales the 4 output accumulators by per-query alpha and returns the
// probability tile already in f16 A-operand layout for the P @ V WMMA.
// ---------------------------------------------------------------------------
__device__ __forceinline__ v16h online_softmax(v8f& s0, v8f& s1,
                                               float& mrow, float& lrow,
                                               v8f* o, int hi) {
  const float scale = 0.125f;               // 1/sqrt(64)
  float tmax = -3.0e38f;
#pragma unroll
  for (int i = 0; i < 8; ++i) {
    s0[i] *= scale; s1[i] *= scale;
    tmax = fmaxf(tmax, fmaxf(s0[i], s1[i]));
  }
  tmax = fmaxf(tmax, __shfl_xor(tmax, 16, 32));   // lanes q <-> q+16
  const float mnew  = fmaxf(mrow, tmax);
  const float alpha = __expf(mrow - mnew);        // first iter: exp(-huge)=0
  mrow = mnew;

  float psum = 0.0f;
  v16h pa;
#pragma unroll
  for (int i = 0; i < 8; ++i) {
    const float p0 = __expf(s0[i] - mnew);
    const float p1 = __expf(s1[i] - mnew);
    psum += p0 + p1;
    pa[i]     = (_Float16)p0;                     // keys koff+i     (tile 0)
    pa[i + 8] = (_Float16)p1;                     // keys 16+koff+i  (tile 1)
  }
  psum += __shfl_xor(psum, 16, 32);
  lrow = lrow * alpha + psum;

  // O-tile VGPR v belongs to query v+8*hi -> fetch its alpha via bpermute.
#pragma unroll
  for (int v = 0; v < 8; ++v) {
    const float av = __shfl(alpha, 8 * hi + v, 32);
    o[0][v] *= av; o[1][v] *= av; o[2][v] *= av; o[3][v] *= av;
  }
  return pa;
}

// ---------------------------------------------------------------------------
// Flash attention, one wave per (batch, head, 32-query block).
// Scores are computed TRANSPOSED: C = K_blk @ Q_blk^T  (lane = query,
// VGPR = key). After exp(), that register layout is exactly the f16
// A-operand layout for the P @ V WMMA -> no LDS transpose needed.
// Two 16-query groups per wave reuse every K-tile and V-tile load.
// Softmax for group A runs before group-C scores to keep live ranges short.
// __launch_bounds__(32): 1-wave workgroup -> RA gets the full register file.
// ---------------------------------------------------------------------------
__global__ void __launch_bounds__(32)
flash_attn_kernel(const _Float16* __restrict__ Q,
                  const _Float16* __restrict__ K,
                  const _Float16* __restrict__ Vt,
                  _Float16* __restrict__ O) {
  const int lane  = threadIdx.x;
  const int bh    = blockIdx.x;        // b*16 + h
  const int b     = bh >> 4;
  const int h     = bh & (NHEAD - 1);
  const int qbase = blockIdx.y * 32;
  const int hi    = lane >> 4;

  const _Float16* Qh = Q  + (size_t)b * SEQ * D_MODEL + h * DK;
  const _Float16* Kh = K  + (size_t)b * SEQ * D_MODEL + h * DK;
  const _Float16* Vh = Vt + (size_t)bh * DK * SEQ;

  // Q tiles as B-operands (N = 16 queries, K = d_k), invariant over key loop.
  const v16h qa0 = load_tile_nt(Qh, qbase,      D_MODEL, 0,  lane);
  const v16h qa1 = load_tile_nt(Qh, qbase,      D_MODEL, 32, lane);
  const v16h qc0 = load_tile_nt(Qh, qbase + 16, D_MODEL, 0,  lane);
  const v16h qc1 = load_tile_nt(Qh, qbase + 16, D_MODEL, 32, lane);

  v8f oa[4] = {}, oc[4] = {};               // 2 x (16 queries x 64 dims)
  float ma = -3.0e38f, la = 0.0f;           // softmax state, group A
  float mc = -3.0e38f, lc = 0.0f;           // softmax state, group C

  for (int kb = 0; kb < SEQ; kb += 32) {
    // Shared K-tile loads (keys kb..kb+31, d_k split 0..31 / 32..63).
    const v16h k00 = load_tile_nt(Kh, kb,      D_MODEL, 0,  lane);
    const v16h k01 = load_tile_nt(Kh, kb,      D_MODEL, 32, lane);
    const v16h k10 = load_tile_nt(Kh, kb + 16, D_MODEL, 0,  lane);
    const v16h k11 = load_tile_nt(Kh, kb + 16, D_MODEL, 32, lane);

    // Group A: scores then softmax (kills sa* before sc* goes live).
    v8f sa0 = {}, sa1 = {};
    sa0 = wmma_f16(k00, qa0, sa0);  sa0 = wmma_f16(k01, qa1, sa0);
    sa1 = wmma_f16(k10, qa0, sa1);  sa1 = wmma_f16(k11, qa1, sa1);
    const v16h paA = online_softmax(sa0, sa1, ma, la, oa, hi);

    // Group C: scores then softmax (K-tiles die here).
    v8f sc0 = {}, sc1 = {};
    sc0 = wmma_f16(k00, qc0, sc0);  sc0 = wmma_f16(k01, qc1, sc0);
    sc1 = wmma_f16(k10, qc0, sc1);  sc1 = wmma_f16(k11, qc1, sc1);
    const v16h paC = online_softmax(sc0, sc1, mc, lc, oc, hi);

    // O += P @ V ; V-tiles shared between the two query groups.
#pragma unroll
    for (int t = 0; t < 4; ++t) {
      const v16h vb = load_tile_nt(Vh, 16 * t, SEQ, kb, lane);
      oa[t] = wmma_f16(paA, vb, oa[t]);
      oc[t] = wmma_f16(paC, vb, oc[t]);
    }
  }

  const float lia = 1.0f / la;
  const float lic = 1.0f / lc;
  const int ln = lane & 15;
#pragma unroll
  for (int v = 0; v < 8; ++v) {
    const float la_v = __shfl(lia, 8 * hi + v, 32);
    const float lc_v = __shfl(lic, 8 * hi + v, 32);
    const int   mrow = v + 8 * hi;
    _Float16* rowA = O + (size_t)(b * SEQ + qbase + mrow) * D_MODEL + h * DK;
    _Float16* rowC = rowA + (size_t)16 * D_MODEL;
#pragma unroll
    for (int t = 0; t < 4; ++t) {
      rowA[16 * t + ln] = (_Float16)(oa[t][v] * la_v);
      rowC[16 * t + ln] = (_Float16)(oc[t][v] * lc_v);
    }
  }
}

// ---------------------------------------------------------------------------
// Host-side orchestration. Workspace layout (f16 elements), total 64 MB.
// ---------------------------------------------------------------------------
extern "C" void kernel_launch(void* const* d_in, const int* in_sizes, int n_in,
                              void* d_out, int out_size, void* d_ws, size_t ws_size,
                              hipStream_t stream) {
  (void)in_sizes; (void)n_in; (void)out_size; (void)ws_size;

  const float* x_q = (const float*)d_in[0];
  const float* x_k = (const float*)d_in[1];
  const float* x_v = (const float*)d_in[2];
  const float* Wq  = (const float*)d_in[3];
  const float* bq  = (const float*)d_in[4];
  const float* Wk  = (const float*)d_in[5];
  const float* bk  = (const float*)d_in[6];
  const float* Wv  = (const float*)d_in[7];
  const float* bv  = (const float*)d_in[8];
  const float* Wo  = (const float*)d_in[9];
  const float* bo  = (const float*)d_in[10];

  const size_t NX = (size_t)BATCH * SEQ * D_MODEL;  // 4Mi elements
  const size_t NW = (size_t)D_MODEL * D_MODEL;      // 1Mi elements

  _Float16* ws   = (_Float16*)d_ws;
  _Float16* xq16 = ws;
  _Float16* xk16 = xq16 + NX;
  _Float16* xv16 = xk16 + NX;
  _Float16* wq16 = xv16 + NX;
  _Float16* wk16 = wq16 + NW;
  _Float16* wv16 = wk16 + NW;
  _Float16* wo16 = wv16 + NW;
  _Float16* q16  = wo16 + NW;
  _Float16* k16  = q16  + NX;
  _Float16* vt16 = k16  + NX;   // [b][h][d][s]
  _Float16* o16  = vt16 + NX;

  // f32 -> f16 casts
  const int n4x = (int)(NX / 4), n4w = (int)(NW / 4);
  dim3 cblk(256);
  cvt_f32_to_f16_kernel<<<dim3((n4x + 255) / 256), cblk, 0, stream>>>(x_q, xq16, n4x);
  cvt_f32_to_f16_kernel<<<dim3((n4x + 255) / 256), cblk, 0, stream>>>(x_k, xk16, n4x);
  cvt_f32_to_f16_kernel<<<dim3((n4x + 255) / 256), cblk, 0, stream>>>(x_v, xv16, n4x);
  cvt_f32_to_f16_kernel<<<dim3((n4w + 255) / 256), cblk, 0, stream>>>(Wq, wq16, n4w);
  cvt_f32_to_f16_kernel<<<dim3((n4w + 255) / 256), cblk, 0, stream>>>(Wk, wk16, n4w);
  cvt_f32_to_f16_kernel<<<dim3((n4w + 255) / 256), cblk, 0, stream>>>(Wv, wv16, n4w);
  cvt_f32_to_f16_kernel<<<dim3((n4w + 255) / 256), cblk, 0, stream>>>(Wo, wo16, n4w);

  // Projections (WMMA GEMMs): grid = (M/32, N/512), 8 waves/block.
  dim3 gblk(256);
  dim3 ggrid((BATCH * SEQ) / 32, D_MODEL / 512);
  gemm_nt_kernel<0><<<ggrid, gblk, 0, stream>>>(xq16, wq16, bq, q16);
  gemm_nt_kernel<0><<<ggrid, gblk, 0, stream>>>(xk16, wk16, bk, k16);
  gemm_nt_kernel<1><<<ggrid, gblk, 0, stream>>>(xv16, wv16, bv, vt16);

  // Flash attention: one wave per (b,h, 32-query block).
  flash_attn_kernel<<<dim3(BATCH * NHEAD, SEQ / 32), dim3(32), 0, stream>>>(
      q16, k16, vt16, o16);

  // Output projection -> f32 d_out
  gemm_nt_kernel<2><<<ggrid, gblk, 0, stream>>>(o16, wo16, bo, (float*)d_out);
}
